// FeatureAdaption_40819369181788
// MI455X (gfx1250) — compile-verified, matrix operand order checked by reading, not run on hardware
//
#include <hip/hip_runtime.h>
#include <hip/hip_bf16.h>

typedef __attribute__((ext_vector_type(16))) __bf16 v16bf;
typedef __attribute__((ext_vector_type(8)))  float  v8f;

#define C_IN   256
#define O_OUT  256
#define HW     4096
#define K2     9
#define KTOT   2304
#define NCHUNK 72          // KTOT / 32
#define NPOS   64          // spatial positions per block (one image row)

__device__ __forceinline__ unsigned pack_bf16(float a, float b) {
    union { __bf16 h[2]; unsigned u; } r;
    r.h[0] = (__bf16)a;
    r.h[1] = (__bf16)b;
    return r.u;
}

// ---------------------------------------------------------------------------
// Pack f32 weight [O][C][3][3] into bf16 A-fragments, layout:
//   [chunk q][mtile t][lane l][8 dwords], K-dim ordered (kpoint*256 + c)
// Per ISA 16-bit A 16x32 layout: lane&15 = M row; lanes<16 -> kbase 0,
// lanes>=16 -> kbase 8; VGPR v<4 holds K = kbase+2v{,+1}; v>=4 holds
// K = 16+kbase+2(v-4){,+1}.
// ---------------------------------------------------------------------------
__global__ void pack_weights_kernel(const float* __restrict__ w,
                                    __bf16* __restrict__ wp) {
    int d = blockIdx.x * blockDim.x + threadIdx.x;          // dword id
    if (d >= NCHUNK * 16 * 32 * 8) return;
    int v = d & 7;
    int l = (d >> 3) & 31;
    int t = (d >> 8) & 15;
    int q = d >> 12;
    int m = t * 16 + (l & 15);
    int kbase = (l < 16) ? 0 : 8;
#pragma unroll
    for (int j = 0; j < 2; ++j) {
        int kic = (v < 4) ? (kbase + v * 2 + j)
                          : (16 + kbase + (v - 4) * 2 + j);
        int r  = q * 32 + kic;          // reduction index = kp*256 + c
        int kp = r >> 8;
        int c  = r & 255;
        wp[d * 2 + j] = (__bf16)w[(m * C_IN + c) * K2 + kp];
    }
}

// ---------------------------------------------------------------------------
// Main DCN implicit-GEMM kernel. One block = one image row (64 positions).
// ---------------------------------------------------------------------------
__global__ __launch_bounds__(256)
void dcn_wmma_kernel(const float* __restrict__ x,
                     const float* __restrict__ sp,
                     const __bf16* __restrict__ wp,
                     float* __restrict__ out) {
    __shared__ int   s_idx[NPOS * K2][4];                 // clamped corner idx
    __shared__ float s_wgt[NPOS * K2][4];                 // validity-masked weights
    __shared__ __align__(32) unsigned sS[2][4][32][8];    // double-buffered B-fragments

    const int tid  = threadIdx.x;
    const int lane = tid & 31;
    const int wv   = tid >> 5;
    const int tilebase = blockIdx.x * NPOS;
    const int b      = tilebase >> 12;          // HW = 4096 per batch
    const int hwbase = tilebase & (HW - 1);
    const int h      = hwbase >> 6;

    // ---- Stage 1: bilinear metadata (base offsets cancel: coord = p0 + sp) ----
    for (int e = tid; e < NPOS * K2; e += 256) {
        int n  = e / K2;
        int kp = e - n * K2;
        float py = (float)h + sp[((b * 18 + 2 * kp) * HW) + hwbase + n];
        float px = (float)n + sp[((b * 18 + 2 * kp + 1) * HW) + hwbase + n];
        float fy = floorf(py), fx = floorf(px);
        int   y0 = (int)fy,    x0 = (int)fx;
        float ly = py - fy,    lx = px - fx;
        float wts[4] = {(1.f - ly) * (1.f - lx), (1.f - ly) * lx,
                        ly * (1.f - lx),         ly * lx};
#pragma unroll
        for (int cidx = 0; cidx < 4; ++cidx) {
            int  yy = y0 + (cidx >> 1);
            int  xx = x0 + (cidx & 1);
            bool valid = (yy >= 0) && (yy < 64) && (xx >= 0) && (xx < 64);
            int  yc = min(max(yy, 0), 63);
            int  xc = min(max(xx, 0), 63);
            s_idx[e][cidx] = yc * 64 + xc;
            s_wgt[e][cidx] = valid ? wts[cidx] : 0.0f;
        }
    }
    __syncthreads();

    v8f acc[2][4];
#pragma unroll
    for (int mi = 0; mi < 2; ++mi)
#pragma unroll
        for (int nt = 0; nt < 4; ++nt)
#pragma unroll
            for (int i = 0; i < 8; ++i) acc[mi][nt][i] = 0.0f;

    const int mt0 = wv * 2;

    // ---- Stage 2: K-loop over 72 chunks of 32 ----
    for (int q = 0; q < NCHUNK; ++q) {
        const int buf   = q & 1;
        const int kp    = q >> 3;
        const int cbase = (q & 7) << 5;

        // gather/lerp 32x64 S-tile straight into B-fragment layout (4 dwords/thread)
#pragma unroll
        for (int i = 0; i < 4; ++i) {
            int sd = tid + i * 256;
            int v  = sd & 7;
            int l  = (sd >> 3) & 31;
            int nt = sd >> 8;
            int n  = nt * 16 + (l & 15);
            int kbase = (l < 16) ? 0 : 8;
            int k0 = (v < 4) ? (kbase + v * 2) : (16 + kbase + (v - 4) * 2);
            int c0 = cbase + k0;
            int e  = n * K2 + kp;
            int   i0 = s_idx[e][0], i1 = s_idx[e][1], i2 = s_idx[e][2], i3 = s_idx[e][3];
            float w0 = s_wgt[e][0], w1 = s_wgt[e][1], w2 = s_wgt[e][2], w3 = s_wgt[e][3];
            const float* xb0 = x + ((b * C_IN + c0) * HW);
            const float* xb1 = xb0 + HW;
            float a0 = w0 * xb0[i0] + w1 * xb0[i1] + w2 * xb0[i2] + w3 * xb0[i3];
            float a1 = w0 * xb1[i0] + w1 * xb1[i1] + w2 * xb1[i2] + w3 * xb1[i3];
            sS[buf][nt][l][v] = pack_bf16(a0, a1);
        }
        __syncthreads();

        v16bf afrag[2];
#pragma unroll
        for (int mi = 0; mi < 2; ++mi)
            afrag[mi] = *(const v16bf*)(wp + ((size_t)((q * 16 + mt0 + mi) * 32 + lane) * 16));

#pragma unroll
        for (int nt = 0; nt < 4; ++nt) {
            v16bf bfrag = *(const v16bf*)(&sS[buf][nt][lane][0]);
#pragma unroll
            for (int mi = 0; mi < 2; ++mi)
                acc[mi][nt] = __builtin_amdgcn_wmma_f32_16x16x32_bf16(
                    false, afrag[mi], false, bfrag,
                    (short)0, acc[mi][nt], false, false);
        }
    }

    // ---- Stage 3: ReLU epilogue, scatter per C/D layout ----
#pragma unroll
    for (int mi = 0; mi < 2; ++mi) {
        int mtile = (mt0 + mi) * 16;
        int madd  = (lane < 16) ? 0 : 8;
#pragma unroll
        for (int nt = 0; nt < 4; ++nt) {
            int n = nt * 16 + (lane & 15);
            float* op = out + ((size_t)(b * O_OUT + mtile + madd) * HW) + hwbase + n;
#pragma unroll
            for (int v = 0; v < 8; ++v) {
                float val = acc[mi][nt][v];
                op[(size_t)v * HW] = val > 0.0f ? val : 0.0f;
            }
        }
    }
}

extern "C" void kernel_launch(void* const* d_in, const int* in_sizes, int n_in,
                              void* d_out, int out_size, void* d_ws, size_t ws_size,
                              hipStream_t stream) {
    const float* x  = (const float*)d_in[0];   // [4,256,64,64]
    const float* sp = (const float*)d_in[1];   // [4,18,64,64]
    const float* w  = (const float*)d_in[2];   // [256,256,3,3]
    float* out = (float*)d_out;                // [4,256,64,64]
    __bf16* wp = (__bf16*)d_ws;                // 1.18 MB packed bf16 weights

    const int ndw = NCHUNK * 16 * 32 * 8;      // 294912 dwords
    pack_weights_kernel<<<(ndw + 255) / 256, 256, 0, stream>>>(w, wp);
    dcn_wmma_kernel<<<256, 256, 0, stream>>>(x, sp, wp, out);
}
